// Block_31396210934364
// MI455X (gfx1250) — matrix operand order repaired and checked by reading, not compiled, not run on hardware
//
#include <hip/hip_runtime.h>
#include <hip/hip_bf16.h>
#include <math.h>

#define B_ 32
#define N_ 577
#define NS_ 576
#define C_ 768

typedef __attribute__((ext_vector_type(16))) __bf16 v16bf;
typedef __attribute__((ext_vector_type(8)))  float  v8f;

__device__ __forceinline__ unsigned short f2bf(float f) {
    unsigned int u = __float_as_uint(f);
    unsigned int r = u + 0x7FFFu + ((u >> 16) & 1u);
    return (unsigned short)(r >> 16);
}

// A-matrix 16x32 bf16 fragment (ISA 7.12.2): lanes0-15 M=lane K{0..7,16..23},
// lanes16-31 M=lane-16 K{8..15,24..31}. Two contiguous 16B chunks per lane.
__device__ __forceinline__ v16bf load_a_frag(const unsigned short* base, int ld,
                                             int row, int k0, int lane) {
    int kbase = k0 + ((lane & 16) ? 8 : 0);
    const unsigned short* p = base + (size_t)row * ld + kbase;
    union { v16bf bf; uint4 u4[2]; } r;
    r.u4[0] = *(const uint4*)(p);
    r.u4[1] = *(const uint4*)(p + 16);
    return r.bf;
}

// B-matrix 32x16 bf16 fragment: lanes0-15 N=lane K{0..7,16..23}, lanes16-31 N=lane-16
// K{8..15,24..31}. Operand stored N-major (ld over K) -> two 16B contiguous loads.
__device__ __forceinline__ v16bf load_b_frag(const unsigned short* base, int ld,
                                             int n, int k0, int lane) {
    int koff = k0 + ((lane & 16) ? 8 : 0);
    const unsigned short* p = base + (size_t)n * ld + koff;
    union { v16bf bf; uint4 u4[2]; } r;
    r.u4[0] = *(const uint4*)(p);
    r.u4[1] = *(const uint4*)(p + 16);
    return r.bf;
}

#define WMMA_BF16(A, Bf, Cacc) \
    __builtin_amdgcn_wmma_f32_16x16x32_bf16(false, (A), false, (Bf), (short)0, (Cacc), false, false)

// ---------------- K1: residual add + LayerNorm ----------------
__global__ void add_ln_kernel(const float* __restrict__ hid, const float* __restrict__ res,
                              const float* __restrict__ nw, const float* __restrict__ nb,
                              float* __restrict__ res_out, float* __restrict__ hs) {
    int row = blockIdx.x;                      // b*N_ + n
    const float* h = hid + (size_t)row * C_;
    const float* r = res + (size_t)row * C_;
    float* ro = res_out + (size_t)row * C_;
    float* o  = hs + (size_t)row * C_;
    float v[3]; float s = 0.f;
#pragma unroll
    for (int i = 0; i < 3; i++) {
        int c = threadIdx.x + i * 256;
        v[i] = h[c] + r[c];
        ro[c] = v[i];
        s += v[i];
    }
    __shared__ float red[256];
    red[threadIdx.x] = s; __syncthreads();
    for (int off = 128; off; off >>= 1) {
        if (threadIdx.x < off) red[threadIdx.x] += red[threadIdx.x + off];
        __syncthreads();
    }
    float mu = red[0] * (1.f / C_);
    __syncthreads();
    float s2 = 0.f;
#pragma unroll
    for (int i = 0; i < 3; i++) { float d = v[i] - mu; s2 += d * d; }
    red[threadIdx.x] = s2; __syncthreads();
    for (int off = 128; off; off >>= 1) {
        if (threadIdx.x < off) red[threadIdx.x] += red[threadIdx.x + off];
        __syncthreads();
    }
    float rs = rsqrtf(red[0] * (1.f / C_) + 1e-5f);
#pragma unroll
    for (int i = 0; i < 3; i++) {
        int c = threadIdx.x + i * 256;
        o[c] = (v[i] - mu) * rs * nw[c] + nb[c];
    }
}

// ---------------- K2: predictor logits + gumbel hard keep ----------------
__global__ void pred_kernel(const float* __restrict__ hs, const float* __restrict__ early,
                            const float* __restrict__ pw, const float* __restrict__ pb,
                            const float* __restrict__ gu, const float* __restrict__ prev,
                            float* __restrict__ hard_keep_out, float* __restrict__ keep,
                            float* __restrict__ pruned) {
    int tok = blockIdx.x * 8 + (threadIdx.x >> 5);
    int lane = threadIdx.x & 31;
    if (tok >= B_ * NS_) return;
    int b = tok / NS_, n = tok % NS_;
    const float* hrow = hs    + ((size_t)(b * N_ + 1 + n)) * C_;
    const float* hcls = hs    + ((size_t)(b * N_)) * C_;
    const float* erow = early + ((size_t)(b * N_ + 1 + n)) * C_;
    const float* ecls = early + ((size_t)(b * N_)) * C_;
    float s0 = 0.f, s1 = 0.f;
    for (int c = lane; c < C_; c += 32) {
        float f = hrow[c] + erow[c] + hcls[c] + ecls[c];
        s0 += f * pw[c * 2 + 0];
        s1 += f * pw[c * 2 + 1];
    }
    for (int off = 16; off; off >>= 1) {
        s0 += __shfl_down(s0, off, 32);
        s1 += __shfl_down(s1, off, 32);
    }
    if (lane == 0) {
        float u0 = gu[(size_t)tok * 2 + 0], u1 = gu[(size_t)tok * 2 + 1];
        float g0 = -logf(-logf(u0)), g1 = -logf(-logf(u1));
        float l0 = s0 + pb[0] + g0, l1 = s1 + pb[1] + g1;
        float y0 = (l0 >= l1) ? 1.f : 0.f;   // argmax (tie -> index 0)
        float pv = prev[tok];
        float hk = y0 * pv;
        hard_keep_out[tok] = hk;
        keep[tok] = hk;
        pruned[tok] = (1.f - hk) * pv;
    }
}

// ---------------- K3: per-batch sums ----------------
__global__ void batch_sum_kernel(const float* __restrict__ keep, const float* __restrict__ pruned,
                                 float* __restrict__ ksum, float* __restrict__ psum,
                                 float* __restrict__ simacc) {
    int b = blockIdx.x;
    float sk = 0.f, sp = 0.f;
    for (int n = threadIdx.x; n < NS_; n += 256) {
        sk += keep[b * NS_ + n];
        sp += pruned[b * NS_ + n];
    }
    __shared__ float rk[256], rp[256];
    rk[threadIdx.x] = sk; rp[threadIdx.x] = sp; __syncthreads();
    for (int off = 128; off; off >>= 1) {
        if (threadIdx.x < off) { rk[threadIdx.x] += rk[threadIdx.x + off];
                                 rp[threadIdx.x] += rp[threadIdx.x + off]; }
        __syncthreads();
    }
    if (threadIdx.x == 0) { ksum[b] = rk[0]; psum[b] = rp[0]; simacc[b] = 0.f; }
}

// ---------------- K4: tp = min_b(int(keep_sum)) / 2 ----------------
__global__ void tp_kernel(const float* __restrict__ ksum, int* __restrict__ tp) {
    int lane = threadIdx.x;
    int v = (lane < B_) ? (int)(ksum[lane] + 0.5f) : 0x7FFFFFFF;
    for (int off = 16; off; off >>= 1) {
        int o = __shfl_down(v, off, 32);
        v = (o < v) ? o : v;
    }
    if (lane == 0) tp[0] = v / 2;
}

// ---------------- K5: stable-partition ranks -> gather/scatter maps --------
__global__ void rank_kernel(const float* __restrict__ keep, const int* __restrict__ tp_ptr,
                            int* __restrict__ seq_src, int* __restrict__ orow) {
    int b = blockIdx.x;
    int n = threadIdx.x;                     // 576 threads
    int tp = tp_ptr[0];
    __shared__ int s[NS_];
    int k = (keep[b * NS_ + n] > 0.5f) ? 1 : 0;
    s[n] = k; __syncthreads();
    for (int off = 1; off < NS_; off <<= 1) {       // inclusive Hillis-Steele
        int t = (n >= off) ? s[n - off] : 0;
        __syncthreads();
        s[n] += t;
        __syncthreads();
    }
    int K = s[NS_ - 1];
    int se = s[n] - k;                              // exclusive prefix of keeps
    int rank = k ? se : (K + (n - se));             // stable partition rank
    int pos = (rank < tp) ? rank : rank + 1;        // skip cls slot at tp
    seq_src[b * N_ + pos] = n;
    orow[b * N_ + pos] = 1 + n;
    if (n == 0) { seq_src[b * N_ + tp] = -1; orow[b * N_ + tp] = 0; }
}

// ---------------- K6: build mixer input rows (bf16) ----------------
__global__ void pack_seq_kernel(const float* __restrict__ hs, const float* __restrict__ keep,
                                const int* __restrict__ seq_src, unsigned short* __restrict__ seq) {
    int row = blockIdx.x;                   // b*N_ + pos
    int b = row / N_;
    int src = seq_src[row];
    const float* base; float scale;
    if (src < 0) { base = hs + (size_t)(b * N_) * C_; scale = 1.f; }
    else         { base = hs + (size_t)(b * N_ + 1 + src) * C_; scale = keep[b * NS_ + src]; }
    unsigned short* o = seq + (size_t)row * C_;
#pragma unroll
    for (int i = 0; i < 3; i++) {
        int c = threadIdx.x + i * 256;
        o[c] = f2bf(base[c] * scale);
    }
}

// ---------------- K7: transpose mixer_w -> bf16 (d-major) ----------------
__global__ void transpose_w_kernel(const float* __restrict__ w, unsigned short* __restrict__ wt) {
    int i = blockIdx.x * 256 + threadIdx.x;      // i = d*C_ + c
    int d = i / C_, c = i % C_;
    wt[i] = f2bf(w[(size_t)c * C_ + d]);
}

// ---------------- K8: normalized spatial rows (bf16) ----------------
__global__ void pack_xn_kernel(const float* __restrict__ hs, unsigned short* __restrict__ xn) {
    int row = blockIdx.x;                        // b*NS_ + n
    int b = row / NS_, n = row % NS_;
    const float* src = hs + (size_t)(b * N_ + 1 + n) * C_;
    float v[3]; float s = 0.f;
#pragma unroll
    for (int i = 0; i < 3; i++) {
        int c = threadIdx.x + i * 256;
        v[i] = src[c]; s += v[i] * v[i];
    }
    __shared__ float red[256];
    red[threadIdx.x] = s; __syncthreads();
    for (int off = 128; off; off >>= 1) {
        if (threadIdx.x < off) red[threadIdx.x] += red[threadIdx.x + off];
        __syncthreads();
    }
    float scale = 1.f / (sqrtf(red[0]) + 1e-6f);
    unsigned short* o = xn + (size_t)row * C_;
#pragma unroll
    for (int i = 0; i < 3; i++) {
        int c = threadIdx.x + i * 256;
        o[c] = f2bf(v[i] * scale);
    }
}

// ------- K9: mixer GEMM, 32x32 tile/wave (2x2 WMMA blocking) + scatter -----
__global__ void mixer_gemm_kernel(const unsigned short* __restrict__ seq,
                                  const unsigned short* __restrict__ wt,
                                  const float* __restrict__ bias,
                                  const int* __restrict__ orow,
                                  float* __restrict__ out) {
    const int TM2 = 19, TN2 = 24;                 // ceil(577/32) x (768/32)
    int wid = blockIdx.x * 8 + (threadIdx.x >> 5);
    int lane = threadIdx.x & 31;
    int b = wid / (TM2 * TN2);
    if (b >= B_) return;
    int t = wid % (TM2 * TN2);
    int tm = t / TN2, tn = t % TN2;
    int m0 = tm * 32, n0 = tn * 32;
    const unsigned short* A = seq + (size_t)b * N_ * C_;
    int ar0 = m0 + (lane & 15);      if (ar0 > N_ - 1) ar0 = N_ - 1;   // pad clamp
    int ar1 = m0 + 16 + (lane & 15); if (ar1 > N_ - 1) ar1 = N_ - 1;
    int br0 = n0 + (lane & 15);
    int br1 = n0 + 16 + (lane & 15);
    v8f acc00 = {}, acc01 = {}, acc10 = {}, acc11 = {};
    for (int k0 = 0; k0 < C_; k0 += 32) {
        v16bf a0 = load_a_frag(A,  C_, ar0, k0, lane);
        v16bf a1 = load_a_frag(A,  C_, ar1, k0, lane);
        v16bf b0 = load_b_frag(wt, C_, br0, k0, lane);
        v16bf b1 = load_b_frag(wt, C_, br1, k0, lane);
        acc00 = WMMA_BF16(a0, b0, acc00);
        acc01 = WMMA_BF16(a0, b1, acc01);
        acc10 = WMMA_BF16(a1, b0, acc10);
        acc11 = WMMA_BF16(a1, b1, acc11);
    }
    int msub = (lane & 16) ? 8 : 0;
    int nn0 = n0 + (lane & 15), nn1 = n0 + 16 + (lane & 15);
    float bv0 = bias[nn0], bv1 = bias[nn1];
    const int* orb = orow + b * N_;
    float* ob = out + (size_t)b * N_ * C_;
#pragma unroll
    for (int v = 0; v < 8; v++) {
        int mA = m0 + msub + v;
        if (mA < N_) {
            int r = orb[mA];
            ob[(size_t)r * C_ + nn0] = acc00[v] + bv0;
            ob[(size_t)r * C_ + nn1] = acc01[v] + bv1;
        }
        int mB = m0 + 16 + msub + v;
        if (mB < N_) {
            int r = orb[mB];
            ob[(size_t)r * C_ + nn0] = acc10[v] + bv0;
            ob[(size_t)r * C_ + nn1] = acc11[v] + bv1;
        }
    }
}

// --- K10: similarity GEMM, 32x32 tile/wave, upper-triangular blocks only ---
// sim0 is symmetric; off-diagonal blocks contribute both orientations'
// weighted squares. Diagonal blocks contain both (m,n),(n,m) already.
__global__ void sim_gemm_kernel(const unsigned short* __restrict__ xn,
                                const float* __restrict__ keep,
                                const float* __restrict__ pruned,
                                float* __restrict__ simacc) {
    const int T2 = 18;                            // 576/32
    const int NTRI = T2 * (T2 + 1) / 2;           // 171 blocks
    int wid = blockIdx.x * 8 + (threadIdx.x >> 5);
    int lane = threadIdx.x & 31;
    int b = wid / NTRI;
    if (b >= B_) return;
    int t = wid % NTRI;
    int tm = 0, rem = t;
    while (rem >= (T2 - tm)) { rem -= (T2 - tm); tm++; }   // tm<=tn enumeration
    int tn = tm + rem;
    int m0 = tm * 32, n0 = tn * 32;
    const unsigned short* X = xn + (size_t)b * NS_ * C_;
    int ar0 = m0 + (lane & 15), ar1 = m0 + 16 + (lane & 15);
    int br0 = n0 + (lane & 15), br1 = n0 + 16 + (lane & 15);
    v8f acc00 = {}, acc01 = {}, acc10 = {}, acc11 = {};
    for (int k0 = 0; k0 < C_; k0 += 32) {
        v16bf a0 = load_a_frag(X, C_, ar0, k0, lane);
        v16bf a1 = load_a_frag(X, C_, ar1, k0, lane);
        v16bf b0 = load_b_frag(X, C_, br0, k0, lane);
        v16bf b1 = load_b_frag(X, C_, br1, k0, lane);
        acc00 = WMMA_BF16(a0, b0, acc00);
        acc01 = WMMA_BF16(a0, b1, acc01);
        acc10 = WMMA_BF16(a1, b0, acc10);
        acc11 = WMMA_BF16(a1, b1, acc11);
    }
    const float* kb = keep + b * NS_;
    const float* pb = pruned + b * NS_;
    int msub = (lane & 16) ? 8 : 0;
    int nn0 = n0 + (lane & 15), nn1 = n0 + 16 + (lane & 15);
    float k_n0 = kb[nn0], k_n1 = kb[nn1];
    float p_n0 = pb[nn0], p_n1 = pb[nn1];
    bool offdiag = (tm != tn);
    float s = 0.f;
#pragma unroll
    for (int v = 0; v < 8; v++) {
        int mA = m0 + msub + v;
        int mB = m0 + 16 + msub + v;
        float pA = pb[mA], kA = kb[mA];
        float pB = pb[mB], kB = kb[mB];
        // acc00: (mA, nn0)   acc01: (mA, nn1)
        // acc10: (mB, nn0)   acc11: (mB, nn1)
        {
            float w = (mA == nn0) ? 0.f : pA * k_n0;
            float val = acc00[v] * w; s += val * val;
            if (offdiag) { float w2 = p_n0 * kA; float v2 = acc00[v] * w2; s += v2 * v2; }
        }
        {
            float w = (mA == nn1) ? 0.f : pA * k_n1;
            float val = acc01[v] * w; s += val * val;
            if (offdiag) { float w2 = p_n1 * kA; float v2 = acc01[v] * w2; s += v2 * v2; }
        }
        {
            float w = (mB == nn0) ? 0.f : pB * k_n0;
            float val = acc10[v] * w; s += val * val;
            if (offdiag) { float w2 = p_n0 * kB; float v2 = acc10[v] * w2; s += v2 * v2; }
        }
        {
            float w = (mB == nn1) ? 0.f : pB * k_n1;
            float val = acc11[v] * w; s += val * val;
            if (offdiag) { float w2 = p_n1 * kB; float v2 = acc11[v] * w2; s += v2 * v2; }
        }
    }
    for (int off = 16; off; off >>= 1) s += __shfl_down(s, off, 32);
    if (lane == 0) atomicAdd(&simacc[b], s);
}

// ---------------- K11: final loss ----------------
__global__ void loss_kernel(const float* __restrict__ simacc, const float* __restrict__ ksum,
                            const float* __restrict__ psum, const int* __restrict__ pcount,
                            float* __restrict__ out_loss) {
    int b = threadIdx.x;
    float v = (b < B_) ? simacc[b] / (ksum[b] * psum[b]) : 0.f;
    for (int off = 16; off; off >>= 1) v += __shfl_down(v, off, 32);
    if (b == 0) out_loss[0] = (pcount[0] < 3) ? (v / (float)B_) : 0.f;
}

extern "C" void kernel_launch(void* const* d_in, const int* in_sizes, int n_in,
                              void* d_out, int out_size, void* d_ws, size_t ws_size,
                              hipStream_t stream) {
    const float* hid   = (const float*)d_in[0];
    const float* res   = (const float*)d_in[1];
    const float* early = (const float*)d_in[2];
    const float* prev  = (const float*)d_in[3];
    const float* gu    = (const float*)d_in[4];
    const float* nw    = (const float*)d_in[5];
    const float* nb    = (const float*)d_in[6];
    const float* pw    = (const float*)d_in[7];
    const float* pb    = (const float*)d_in[8];
    const float* mw    = (const float*)d_in[9];
    const float* mb    = (const float*)d_in[10];
    const int*   pcount = (const int*)d_in[11];

    float* out_mixed = (float*)d_out;                                   // (B,N,C)
    float* out_res   = out_mixed + (size_t)B_ * N_ * C_;                // (B,N,C)
    float* out_keep  = out_res   + (size_t)B_ * N_ * C_;                // (B,NS,1)
    float* out_loss  = out_keep  + (size_t)B_ * NS_;                    // scalar

    // workspace carve-up (256B aligned)
    char* ws = (char*)d_ws;
    size_t off = 0;
    auto carve = [&](size_t bytes) -> char* {
        off = (off + 255) & ~(size_t)255;
        char* p = ws + off;
        off += bytes;
        return p;
    };
    float*          hs      = (float*)carve((size_t)B_ * N_ * C_ * 4);
    unsigned short* seq     = (unsigned short*)carve((size_t)B_ * N_ * C_ * 2);
    unsigned short* xn      = (unsigned short*)carve((size_t)B_ * NS_ * C_ * 2);
    unsigned short* wt      = (unsigned short*)carve((size_t)C_ * C_ * 2);
    float*          keep    = (float*)carve((size_t)B_ * NS_ * 4);
    float*          pruned  = (float*)carve((size_t)B_ * NS_ * 4);
    int*            seq_src = (int*)carve((size_t)B_ * N_ * 4);
    int*            orow    = (int*)carve((size_t)B_ * N_ * 4);
    float*          ksum    = (float*)carve(B_ * 4);
    float*          psum    = (float*)carve(B_ * 4);
    float*          simacc  = (float*)carve(B_ * 4);
    int*            tp      = (int*)carve(4);
    (void)ws_size; (void)in_sizes; (void)n_in; (void)out_size;

    add_ln_kernel<<<B_ * N_, 256, 0, stream>>>(hid, res, nw, nb, out_res, hs);
    pred_kernel<<<(B_ * NS_ + 7) / 8, 256, 0, stream>>>(hs, early, pw, pb, gu, prev,
                                                        out_keep, keep, pruned);
    batch_sum_kernel<<<B_, 256, 0, stream>>>(keep, pruned, ksum, psum, simacc);
    tp_kernel<<<1, 32, 0, stream>>>(ksum, tp);
    rank_kernel<<<B_, NS_, 0, stream>>>(keep, tp, seq_src, orow);
    pack_seq_kernel<<<B_ * N_, 256, 0, stream>>>(hs, keep, seq_src, seq);
    transpose_w_kernel<<<(C_ * C_) / 256, 256, 0, stream>>>(mw, wt);
    pack_xn_kernel<<<B_ * NS_, 256, 0, stream>>>(hs, xn);
    mixer_gemm_kernel<<<(B_ * 19 * 24 + 7) / 8, 256, 0, stream>>>(seq, wt, mb, orow, out_mixed);
    sim_gemm_kernel<<<(B_ * 171 + 7) / 8, 256, 0, stream>>>(xn, keep, pruned, simacc);
    loss_kernel<<<1, 32, 0, stream>>>(simacc, ksum, psum, pcount, out_loss);
}
